// CausalSelfAttention_15504831938904
// MI455X (gfx1250) — matrix operand order repaired and checked by reading, not compiled, not run on hardware
//
#include <hip/hip_runtime.h>

// ---------------------------------------------------------------------------
// Problem constants (from reference): B=2, T=2048, D_MODEL=1024, H=16, Dh=64
// ---------------------------------------------------------------------------
#define BB  2
#define TT  2048
#define CC  1024
#define HH  16
#define DH  64
#define N3  3072          // 3*D_MODEL
#define MAXD 128          // MAX_DIST

typedef __attribute__((ext_vector_type(16))) _Float16 half16;
typedef __attribute__((ext_vector_type(8)))  _Float16 half8;
typedef __attribute__((ext_vector_type(8)))  float    float8;

#define WMMA(a, b, c) \
  __builtin_amdgcn_wmma_f32_16x16x32_f16(false, (a), false, (b), (short)0, (c), false, false)

// ---------------------------------------------------------------------------
// Fragment loaders (wave32). Source operand is row-major [16][ld] tile where
// rows are A-rows (A frag) or B-columns (B frag, i.e. operand stored N-major).
//
// A 16x32 f16 layout (ISA 7.12.2): lanes 0-15 row M=lane, K={0..7,16..23};
// lanes 16-31 row M=lane-16, K={8..15,24..31}.
// B 32x16 f16 layout: lane = column; lanes 0-15 K={0..7,16..23}, lanes 16-31 K={8..15,24..31}.
// ---------------------------------------------------------------------------
__device__ __forceinline__ half16 load_a_frag(const _Float16* tile, int ld, int lane) {
  const _Float16* p = tile + (size_t)(lane & 15) * ld + ((lane >> 4) << 3);
  half8 lo = *(const half8*)(p);
  half8 hi = *(const half8*)(p + 16);
  half16 f;
#pragma unroll
  for (int i = 0; i < 8; ++i) { f[i] = lo[i]; f[i + 8] = hi[i]; }
  return f;
}

__device__ __forceinline__ half16 load_b_frag(const _Float16* tile, int ld, int lane) {
  const _Float16* p = tile + (size_t)(lane & 15) * ld + ((lane >> 4) << 3);
  half8 lo = *(const half8*)(p);
  half8 hi = *(const half8*)(p + 16);
  half16 f;
#pragma unroll
  for (int i = 0; i < 8; ++i) { f[i] = lo[i]; f[i + 8] = hi[i]; }
  return f;
}

// B-fragment with the per-lane column index clamped (causal boundary tiles).
__device__ __forceinline__ half16 load_b_frag_clamp(const _Float16* base, int ld, int lane,
                                                    int col0, int maxcol) {
  int col = col0 + (lane & 15);
  col = col > maxcol ? maxcol : col;
  const _Float16* p = base + (size_t)col * ld + ((lane >> 4) << 3);
  half8 lo = *(const half8*)(p);
  half8 hi = *(const half8*)(p + 16);
  half16 f;
#pragma unroll
  for (int i = 0; i < 8; ++i) { f[i] = lo[i]; f[i + 8] = hi[i]; }
  return f;
}

// ---------------------------------------------------------------------------
// Staging kernels: f32 -> f16 copies / transposes.
// ---------------------------------------------------------------------------
__global__ void k_cvt(const float* __restrict__ in, _Float16* __restrict__ out, int n) {
  int i = blockIdx.x * blockDim.x + threadIdx.x;
  if (i < n) out[i] = (_Float16)in[i];
}

// out[c*R + r] = in[r*C + c]
__global__ void k_transpose_cvt(const float* __restrict__ in, _Float16* __restrict__ out,
                                int R, int C) {
  int i = blockIdx.x * blockDim.x + threadIdx.x;
  if (i >= R * C) return;
  int r = i / C, c = i - r * C;
  out[(size_t)c * R + r] = (_Float16)in[i];
}

// ---------------------------------------------------------------------------
// QKV GEMM: [4096 x 1024] x [1024 x 3072] + bqkv, routed into head layouts:
//   Q -> Qh[b,h,t,d] (pre-scaled by 1/sqrt(Dh)), K -> Kh[b,h,t,d],
//   V -> Vt[b,h,d,t] (transposed for the P*V B-operand).
//
// Register-blocked: each wave computes a 32x64 tile (2 A-frags x 4 B-frags ->
// 8 WMMAs per 12 b128 loads per k-step). 8 waves / block as 2x4 -> 64x256.
// grid (N3/256, M/64).
// ---------------------------------------------------------------------------
__global__ void k_qkv_gemm(const _Float16* __restrict__ X,     // [4096][1024]
                           const _Float16* __restrict__ Wt,    // [3072][1024] (Wqkv^T)
                           const float* __restrict__ bqkv,     // [3072]
                           _Float16* __restrict__ Qh,
                           _Float16* __restrict__ Kh,
                           _Float16* __restrict__ Vt) {
  const int lane = threadIdx.x & 31;
  const int w    = threadIdx.x >> 5;
  const int m0   = blockIdx.y * 64  + (w >> 2) * 32;   // wave M origin (2 tiles)
  const int n0   = blockIdx.x * 256 + (w & 3) * 64;    // wave N origin (4 tiles)

  const _Float16* A0 = X  + (size_t)m0 * CC;
  const _Float16* A1 = X  + (size_t)(m0 + 16) * CC;
  const _Float16* Bt = Wt + (size_t)n0 * CC;

  float8 acc[2][4] = {{{}, {}, {}, {}}, {{}, {}, {}, {}}};
#pragma unroll 2
  for (int k0 = 0; k0 < CC; k0 += 32) {
    __builtin_prefetch(A0 + k0 + 128, 0, 1);
    half16 a0 = load_a_frag(A0 + k0, CC, lane);
    half16 a1 = load_a_frag(A1 + k0, CC, lane);
    half16 b0 = load_b_frag(Bt + (size_t)0  * CC + k0, CC, lane);
    half16 b1 = load_b_frag(Bt + (size_t)16 * CC + k0, CC, lane);
    half16 b2 = load_b_frag(Bt + (size_t)32 * CC + k0, CC, lane);
    half16 b3 = load_b_frag(Bt + (size_t)48 * CC + k0, CC, lane);
    acc[0][0] = WMMA(a0, b0, acc[0][0]);
    acc[0][1] = WMMA(a0, b1, acc[0][1]);
    acc[0][2] = WMMA(a0, b2, acc[0][2]);
    acc[0][3] = WMMA(a0, b3, acc[0][3]);
    acc[1][0] = WMMA(a1, b0, acc[1][0]);
    acc[1][1] = WMMA(a1, b1, acc[1][1]);
    acc[1][2] = WMMA(a1, b2, acc[1][2]);
    acc[1][3] = WMMA(a1, b3, acc[1][3]);
  }

  const int hi = lane >> 4;
  const int ln = lane & 15;
#pragma unroll
  for (int j = 0; j < 4; ++j) {
    const int col = n0 + j * 16 + ln;        // 0..3071; which/h tile-uniform
    const int which = col >> 10;             // 0=Q 1=K 2=V
    const int cc  = col & (CC - 1);
    const int h   = cc >> 6;
    const int d   = cc & (DH - 1);
    const float bias = bqkv[col];
#pragma unroll
    for (int i = 0; i < 2; ++i) {
#pragma unroll
      for (int r = 0; r < 8; ++r) {
        const int row = m0 + i * 16 + r + hi * 8;  // row = b*T + t
        const int bh  = (row >> 11) * HH + h;      // b*H + h  (T == 1<<11)
        const int t   = row & (TT - 1);
        const float v = acc[i][j][r] + bias;
        if (which == 0)
          Qh[((size_t)bh * TT + t) * DH + d] = (_Float16)(v * 0.125f);  // 1/sqrt(64)
        else if (which == 1)
          Kh[((size_t)bh * TT + t) * DH + d] = (_Float16)v;
        else
          Vt[((size_t)bh * DH + d) * TT + t] = (_Float16)v;
      }
    }
  }
}

// ---------------------------------------------------------------------------
// Flash-style causal attention with relative-position bias.
// One wave (32 threads) per (b, h, 16-query tile). grid(T/16, H, B).
// ---------------------------------------------------------------------------
__global__ void k_attn(const _Float16* __restrict__ Qh,
                       const _Float16* __restrict__ Kh,
                       const _Float16* __restrict__ Vt,
                       const float* __restrict__ rel_emb,   // [257][16]
                       _Float16* __restrict__ Ah) {         // [4096][1024]
  __shared__ __align__(16) _Float16 ldsP[16 * 40];          // 80B row stride

  const int lane = threadIdx.x;
  const int q0   = blockIdx.x * 16;
  const int h    = blockIdx.y;
  const int b    = blockIdx.z;
  const int hi   = lane >> 4;
  const int ln   = lane & 15;

  const size_t hb = (size_t)(b * HH + h);
  const _Float16* Q = Qh + hb * TT * DH;
  const _Float16* K = Kh + hb * TT * DH;
  const _Float16* V = Vt + hb * DH * TT;

  const half16 qf0 = load_a_frag(Q + (size_t)q0 * DH +  0, DH, lane);
  const half16 qf1 = load_a_frag(Q + (size_t)q0 * DH + 32, DH, lane);

  float8 o[4] = {{}, {}, {}, {}};
  float mrow[8], lrow[8];
#pragma unroll
  for (int r = 0; r < 8; ++r) { mrow[r] = -1e30f; lrow[r] = 0.0f; }

  for (int s0 = 0; s0 <= q0 + 15; s0 += 32) {
    // ---- S = (Q/sqrt(Dh)) * K^T : two 16x16 tiles covering 32 keys ----
    float8 s[2];
#pragma unroll
    for (int nt = 0; nt < 2; ++nt) {
      half16 kf0 = load_b_frag_clamp(K +  0, DH, lane, s0 + nt * 16, TT - 1);
      half16 kf1 = load_b_frag_clamp(K + 32, DH, lane, s0 + nt * 16, TT - 1);
      float8 acc = {};
      acc = WMMA(qf0, kf0, acc);
      acc = WMMA(qf1, kf1, acc);
      s[nt] = acc;
    }

    // ---- bias + causal mask + online softmax update ----
#pragma unroll
    for (int r = 0; r < 8; ++r) {
      const int t = q0 + r + hi * 8;

      int si0 = s0 + ln;
      int rel0 = si0 - t;
      rel0 = (rel0 < -MAXD ? -MAXD : (rel0 > MAXD ? MAXD : rel0)) + MAXD;
      float v0 = (si0 > t) ? -1e30f : s[0][r] + rel_emb[rel0 * HH + h];

      int si1 = s0 + 16 + ln;
      int rel1 = si1 - t;
      rel1 = (rel1 < -MAXD ? -MAXD : (rel1 > MAXD ? MAXD : rel1)) + MAXD;
      float v1 = (si1 > t) ? -1e30f : s[1][r] + rel_emb[rel1 * HH + h];

      float mx = fmaxf(v0, v1);
#pragma unroll
      for (int off = 8; off >= 1; off >>= 1) mx = fmaxf(mx, __shfl_xor(mx, off));

      const float mnew  = fmaxf(mrow[r], mx);
      const float alpha = __expf(mrow[r] - mnew);
      const float p0 = __expf(v0 - mnew);
      const float p1 = __expf(v1 - mnew);
      float ps = p0 + p1;
#pragma unroll
      for (int off = 8; off >= 1; off >>= 1) ps += __shfl_xor(ps, off);

      lrow[r] = lrow[r] * alpha + ps;
      mrow[r] = mnew;
#pragma unroll
      for (int dt = 0; dt < 4; ++dt) o[dt][r] *= alpha;

      const int m = r + hi * 8;
      ldsP[m * 40 + ln]      = (_Float16)p0;
      ldsP[m * 40 + 16 + ln] = (_Float16)p1;
    }
    __syncthreads();   // single-wave workgroup: compiler fence + ds waits

    // ---- reshape P (C-layout in LDS) into an A-fragment, then O += P*V ----
    half16 pf;
    {
      const _Float16* lp = &ldsP[ln * 40 + hi * 8];
      half8 lo = *(const half8*)(lp);
      half8 hiv = *(const half8*)(lp + 16);
#pragma unroll
      for (int i = 0; i < 8; ++i) { pf[i] = lo[i]; pf[i + 8] = hiv[i]; }
    }
#pragma unroll
    for (int dt = 0; dt < 4; ++dt) {
      half16 vf = load_b_frag(V + (size_t)(dt * 16) * TT + s0, TT, lane);
      o[dt] = WMMA(pf, vf, o[dt]);
    }
    __syncthreads();
  }

  // ---- normalize and store into [B*T][C] activation buffer ----
#pragma unroll
  for (int r = 0; r < 8; ++r) {
    const float inv = 1.0f / lrow[r];
    const int t = q0 + r + hi * 8;
    const size_t row = (size_t)b * TT + t;
#pragma unroll
    for (int dt = 0; dt < 4; ++dt) {
      const int col = h * DH + dt * 16 + ln;
      Ah[row * CC + col] = (_Float16)(o[dt][r] * inv);
    }
  }
}

// ---------------------------------------------------------------------------
// Output projection GEMM: [4096 x 1024] x [1024 x 1024] + bout -> f32 d_out.
// Same 32x64 wave tile / 64x256 block tile as k_qkv_gemm. grid (CC/256, M/64).
// ---------------------------------------------------------------------------
__global__ void k_out_gemm(const _Float16* __restrict__ A,    // [4096][1024]
                           const _Float16* __restrict__ Bt,   // [1024][1024] (Wout^T)
                           const float* __restrict__ bout,    // [1024]
                           float* __restrict__ out) {         // [4096][1024]
  const int lane = threadIdx.x & 31;
  const int w    = threadIdx.x >> 5;
  const int m0   = blockIdx.y * 64  + (w >> 2) * 32;
  const int n0   = blockIdx.x * 256 + (w & 3) * 64;

  const _Float16* A0 = A  + (size_t)m0 * CC;
  const _Float16* A1 = A  + (size_t)(m0 + 16) * CC;
  const _Float16* Bp = Bt + (size_t)n0 * CC;

  float8 acc[2][4] = {{{}, {}, {}, {}}, {{}, {}, {}, {}}};
#pragma unroll 2
  for (int k0 = 0; k0 < CC; k0 += 32) {
    __builtin_prefetch(A0 + k0 + 128, 0, 1);
    half16 a0 = load_a_frag(A0 + k0, CC, lane);
    half16 a1 = load_a_frag(A1 + k0, CC, lane);
    half16 b0 = load_b_frag(Bp + (size_t)0  * CC + k0, CC, lane);
    half16 b1 = load_b_frag(Bp + (size_t)16 * CC + k0, CC, lane);
    half16 b2 = load_b_frag(Bp + (size_t)32 * CC + k0, CC, lane);
    half16 b3 = load_b_frag(Bp + (size_t)48 * CC + k0, CC, lane);
    acc[0][0] = WMMA(a0, b0, acc[0][0]);
    acc[0][1] = WMMA(a0, b1, acc[0][1]);
    acc[0][2] = WMMA(a0, b2, acc[0][2]);
    acc[0][3] = WMMA(a0, b3, acc[0][3]);
    acc[1][0] = WMMA(a1, b0, acc[1][0]);
    acc[1][1] = WMMA(a1, b1, acc[1][1]);
    acc[1][2] = WMMA(a1, b2, acc[1][2]);
    acc[1][3] = WMMA(a1, b3, acc[1][3]);
  }

  const int hi = lane >> 4;
  const int ln = lane & 15;
#pragma unroll
  for (int j = 0; j < 4; ++j) {
    const int col = n0 + j * 16 + ln;
    const float bias = bout[col];
#pragma unroll
    for (int i = 0; i < 2; ++i) {
#pragma unroll
      for (int r = 0; r < 8; ++r) {
        const int row = m0 + i * 16 + r + hi * 8;
        out[(size_t)row * CC + col] = acc[i][j][r] + bias;
      }
    }
  }
}

// ---------------------------------------------------------------------------
// Host-side launcher.
// ---------------------------------------------------------------------------
extern "C" void kernel_launch(void* const* d_in, const int* in_sizes, int n_in,
                              void* d_out, int out_size, void* d_ws, size_t ws_size,
                              hipStream_t stream) {
  const float* x    = (const float*)d_in[0];   // [2,2048,1024]
  const float* Wqkv = (const float*)d_in[1];   // [1024,3072]
  const float* bqkv = (const float*)d_in[2];   // [3072]
  const float* Wout = (const float*)d_in[3];   // [1024,1024]
  const float* bout = (const float*)d_in[4];   // [1024]
  const float* rel  = (const float*)d_in[5];   // [257,16]
  float* out = (float*)d_out;

  char* ws = (char*)d_ws;
  size_t off = 0;
  auto halloc = [&](size_t elems) {
    _Float16* p = (_Float16*)(ws + off);
    off += elems * sizeof(_Float16);
    return p;
  };
  _Float16* xh    = halloc((size_t)BB * TT * CC);   // x in f16
  _Float16* WqkvT = halloc((size_t)N3 * CC);        // Wqkv^T f16
  _Float16* WoutT = halloc((size_t)CC * CC);        // Wout^T f16
  _Float16* Qh    = halloc((size_t)BB * HH * TT * DH);
  _Float16* Kh    = halloc((size_t)BB * HH * TT * DH);
  _Float16* Vt    = halloc((size_t)BB * HH * DH * TT);
  _Float16* Ah    = halloc((size_t)BB * TT * CC);   // attention output, f16
  (void)ws_size;

  const int nx = BB * TT * CC;
  k_cvt<<<(nx + 255) / 256, 256, 0, stream>>>(x, xh, nx);
  k_transpose_cvt<<<(CC * N3 + 255) / 256, 256, 0, stream>>>(Wqkv, WqkvT, CC, N3);
  k_transpose_cvt<<<(CC * CC + 255) / 256, 256, 0, stream>>>(Wout, WoutT, CC, CC);

  k_qkv_gemm<<<dim3(N3 / 256, (BB * TT) / 64), 256, 0, stream>>>(xh, WqkvT, bqkv, Qh, Kh, Vt);

  k_attn<<<dim3(TT / 16, HH, BB), 32, 0, stream>>>(Qh, Kh, Vt, rel, Ah);

  k_out_gemm<<<dim3(CC / 256, (BB * TT) / 64), 256, 0, stream>>>(Ah, WoutT, bout, out);
}
